// RQVAE_71347996721155
// MI455X (gfx1250) — compile-verified, matrix operand order checked by reading, not run on hardware
//
#include <hip/hip_runtime.h>
#include <hip/hip_bf16.h>
#include <math.h>

#define N_TOK  32768
#define DIM    256
#define KCODES 4096
#define NCB    4

typedef __bf16 bf16;
typedef __attribute__((ext_vector_type(16))) bf16  v16bf_t;
typedef __attribute__((ext_vector_type(8)))  bf16  v8bf_t;
typedef __attribute__((ext_vector_type(8)))  float v8f_t;
typedef __attribute__((ext_vector_type(4)))  float v4f_t;

// Load one 16-element bf16 A-fragment (16x16x32 bf16 WMMA layout) from an f32 row.
// off0 = k0 + lo, where lo = (lane<16 ? 0 : 8). Elements 0..7 <- k0+lo..+7,
// elements 8..15 <- k0+16+lo..+7 (ISA 16-bit A-operand layout).
__device__ __forceinline__ v16bf_t load_a_chunk(const float* __restrict__ arow, int off0) {
  const v4f_t* p0 = (const v4f_t*)(arow + off0);
  const v4f_t* p1 = (const v4f_t*)(arow + off0 + 16);
  v4f_t x0 = p0[0], x1 = p0[1], y0 = p1[0], y1 = p1[1];
  v16bf_t r;
#pragma unroll
  for (int j = 0; j < 4; ++j) {
    r[j]      = (bf16)x0[j];
    r[4 + j]  = (bf16)x1[j];
    r[8 + j]  = (bf16)y0[j];
    r[12 + j] = (bf16)y1[j];
  }
  return r;
}

// Load one 16-element bf16 B-fragment (two contiguous 16B loads).
__device__ __forceinline__ v16bf_t load_b_chunk(const bf16* __restrict__ brow, int off0) {
  const v8bf_t* q0 = (const v8bf_t*)(brow + off0);
  const v8bf_t* q1 = (const v8bf_t*)(brow + off0 + 16);
  v8bf_t h0 = *q0, h1 = *q1;
  return __builtin_shufflevector(h0, h1, 0, 1, 2, 3, 4, 5, 6, 7,
                                 8, 9, 10, 11, 12, 13, 14, 15);
}

// 16 WMMAs over one 16-code B tile (two 16-token accumulator tiles) + argmin update.
__device__ __forceinline__ void wmma_step(const v16bf_t a0[8], const v16bf_t a1[8],
                                          const v16bf_t b[8], float cn2, int code,
                                          float bd0[8], int bn0[8],
                                          float bd1[8], int bn1[8]) {
  v8f_t acc0 = {0.f, 0.f, 0.f, 0.f, 0.f, 0.f, 0.f, 0.f};
  v8f_t acc1 = {0.f, 0.f, 0.f, 0.f, 0.f, 0.f, 0.f, 0.f};
#pragma unroll
  for (int c = 0; c < 8; ++c) {
    acc0 = __builtin_amdgcn_wmma_f32_16x16x32_bf16(false, a0[c], false, b[c],
                                                   (short)0, acc0, false, false);
    acc1 = __builtin_amdgcn_wmma_f32_16x16x32_bf16(false, a1[c], false, b[c],
                                                   (short)0, acc1, false, false);
  }
#pragma unroll
  for (int v = 0; v < 8; ++v) {
    float d0 = fmaf(-2.0f, acc0[v], cn2);
    if (d0 < bd0[v]) { bd0[v] = d0; bn0[v] = code; }
    float d1 = fmaf(-2.0f, acc1[v], cn2);
    if (d1 < bd1[v]) { bd1[v] = d1; bn1[v] = code; }
  }
}

// Fused distance-GEMM + argmin. One wave handles 32 tokens (two 16x16 C tiles)
// across all 4096 codes, double-buffering the B (codebook) tiles so the 16
// global_load_b128 of tile t+1 overlap the 16 WMMAs of tile t.
// dist = ||c||^2 - 2*<r,c> (||r||^2 is argmin-invariant).
__global__ void __launch_bounds__(128)
rqvae_argmin_wmma(const float* __restrict__ residual,   // [N_TOK, DIM] f32
                  const bf16*  __restrict__ cb_bf,      // [KCODES, DIM] bf16 (this layer)
                  const float* __restrict__ cb_norm,    // [KCODES] f32 (exact)
                  int* __restrict__ idx_out)            // [N_TOK]
{
  const int lane = threadIdx.x & 31;
  const int wave = threadIdx.x >> 5;
  const int l16  = lane & 15;
  const int half = lane >> 4;          // 0: lanes 0-15, 1: lanes 16-31
  const int lo   = half ? 8 : 0;
  const int tok_base = (blockIdx.x * 4 + wave) * 32;

  // A fragments for two 16-token tiles, all of K=256 (8 chunks of 32).
  v16bf_t a0[8], a1[8];
  const float* ar0 = residual + (size_t)(tok_base + l16) * DIM;
  const float* ar1 = residual + (size_t)(tok_base + 16 + l16) * DIM;
#pragma unroll
  for (int c = 0; c < 8; ++c) {
    a0[c] = load_a_chunk(ar0, c * 32 + lo);
    a1[c] = load_a_chunk(ar1, c * 32 + lo);
  }

  float bd0[8], bd1[8];
  int   bn0[8], bn1[8];
#pragma unroll
  for (int v = 0; v < 8; ++v) { bd0[v] = 3.4e38f; bd1[v] = 3.4e38f; bn0[v] = 0; bn1[v] = 0; }

  // Prologue: load B tile 0 into b0.
  v16bf_t b0[8], b1[8];
  {
    const bf16* brow = cb_bf + (size_t)l16 * DIM;
#pragma unroll
    for (int c = 0; c < 8; ++c) b0[c] = load_b_chunk(brow, c * 32 + lo);
  }
  float cn2a = cb_norm[l16];

  for (int nb = 0; nb < KCODES; nb += 32) {
    // Prefetch tile nb+16 into b1 (nb+16 < KCODES always: nb <= KCODES-32).
    {
      const bf16* brow = cb_bf + (size_t)(nb + 16 + l16) * DIM;
#pragma unroll
      for (int c = 0; c < 8; ++c) b1[c] = load_b_chunk(brow, c * 32 + lo);
    }
    float cn2b = cb_norm[nb + 16 + l16];
    wmma_step(a0, a1, b0, cn2a, nb + l16, bd0, bn0, bd1, bn1);

    // Prefetch tile nb+32 into b0 (wrap to 0 on last iteration; harmless reload).
    {
      const int nbn = (nb + 32) & (KCODES - 1);
      const bf16* brow = cb_bf + (size_t)(nbn + l16) * DIM;
#pragma unroll
      for (int c = 0; c < 8; ++c) b0[c] = load_b_chunk(brow, c * 32 + lo);
      cn2a = cb_norm[nbn + l16];
    }
    wmma_step(a0, a1, b1, cn2b, nb + 16 + l16, bd0, bn0, bd1, bn1);
  }

  // Cross-lane argmin within each 16-lane group (C layout: lanes 0-15 hold M=v,
  // lanes 16-31 hold M=v+8; column N == l16).
#pragma unroll
  for (int v = 0; v < 8; ++v) {
    float d = bd0[v]; int n = bn0[v];
#pragma unroll
    for (int off = 1; off < 16; off <<= 1) {
      float od = __shfl_xor(d, off, 16);
      int   on = __shfl_xor(n, off, 16);
      if (od < d || (od == d && on < n)) { d = od; n = on; }
    }
    if (l16 == 0) idx_out[tok_base + half * 8 + v] = n;

    d = bd1[v]; n = bn1[v];
#pragma unroll
    for (int off = 1; off < 16; off <<= 1) {
      float od = __shfl_xor(d, off, 16);
      int   on = __shfl_xor(n, off, 16);
      if (od < d || (od == d && on < n)) { d = od; n = on; }
    }
    if (l16 == 0) idx_out[tok_base + 16 + half * 8 + v] = n;
  }
}

// Per-layer residual update, straight-through sum, losses, histogram, index output.
__global__ void __launch_bounds__(256)
rqvae_update(const float* __restrict__ cb,          // [KCODES, DIM] f32 (this layer)
             const int*   __restrict__ idx,         // [N_TOK]
             float* __restrict__ residual,          // [N_TOK, DIM] in/out
             float* __restrict__ qsum,              // [N_TOK, DIM] (d_out)
             float* __restrict__ idx_out_f,         // [N_TOK] (d_out, this layer)
             int*   __restrict__ hist,              // [KCODES] (this layer)
             float* __restrict__ tok_sumsq,         // [N_TOK] (this layer)
             int first_layer)
{
  const int t = blockIdx.x;
  const int d = threadIdx.x;
  const int code = idx[t];
  const size_t off = (size_t)t * DIM + d;
  const float q = cb[(size_t)code * DIM + d];
  const float r = residual[off];
  const float diff = r - q;
  residual[off] = diff;
  if (first_layer) qsum[off] = q; else qsum[off] += q;

  float s = diff * diff;
#pragma unroll
  for (int o = 16; o >= 1; o >>= 1) s += __shfl_down(s, o, 32);
  __shared__ float part[8];
  if ((threadIdx.x & 31) == 0) part[threadIdx.x >> 5] = s;
  __syncthreads();
  if (threadIdx.x == 0) {
    float tot = 0.f;
#pragma unroll
    for (int i = 0; i < 8; ++i) tot += part[i];
    tok_sumsq[t] = tot;
    idx_out_f[t] = (float)code;
    atomicAdd(&hist[code], 1);
  }
}

// Convert all codebooks to bf16 and compute exact f32 norms. One block per code.
__global__ void __launch_bounds__(256)
rqvae_prep_cb(const float* __restrict__ cb, bf16* __restrict__ cb_bf,
              float* __restrict__ cb_norm)
{
  const int code = blockIdx.x;                // 0 .. NCB*KCODES-1
  const size_t off = (size_t)code * DIM + threadIdx.x;
  const float v = cb[off];
  cb_bf[off] = (bf16)v;
  float s = v * v;
#pragma unroll
  for (int o = 16; o >= 1; o >>= 1) s += __shfl_down(s, o, 32);
  __shared__ float part[8];
  if ((threadIdx.x & 31) == 0) part[threadIdx.x >> 5] = s;
  __syncthreads();
  if (threadIdx.x == 0) {
    float tot = 0.f;
#pragma unroll
    for (int i = 0; i < 8; ++i) tot += part[i];
    cb_norm[code] = tot;
  }
}

__global__ void rqvae_zero_hist(int* __restrict__ hist) {
  int i = blockIdx.x * blockDim.x + threadIdx.x;
  if (i < NCB * KCODES) hist[i] = 0;
}

// Deterministic loss + perplexity reduction; fully overwrites the scalar outputs.
__global__ void __launch_bounds__(256)
rqvae_finalize(const int* __restrict__ hist,         // [NCB, KCODES]
               const float* __restrict__ tok_sumsq,  // [NCB, N_TOK]
               float* __restrict__ scalars)          // [2]: vq_loss, total_perp
{
  __shared__ float sdata[256];
  const int tid = threadIdx.x;

  float s = 0.f;
  for (int i = tid; i < NCB * N_TOK; i += 256) s += tok_sumsq[i];
  sdata[tid] = s; __syncthreads();
  for (int o = 128; o >= 1; o >>= 1) { if (tid < o) sdata[tid] += sdata[tid + o]; __syncthreads(); }
  float loss = 0.f;
  if (tid == 0) loss = 1.25f * sdata[0] * (1.0f / ((float)N_TOK * (float)DIM));
  __syncthreads();

  float perp = 0.f;
  for (int l = 0; l < NCB; ++l) {
    float e = 0.f;
    for (int i = tid; i < KCODES; i += 256) {
      float p = (float)hist[l * KCODES + i] * (1.0f / (float)N_TOK);
      e += p * logf(p + 1e-10f);
    }
    sdata[tid] = e; __syncthreads();
    for (int o = 128; o >= 1; o >>= 1) { if (tid < o) sdata[tid] += sdata[tid + o]; __syncthreads(); }
    if (tid == 0) perp += expf(-sdata[0]);
    __syncthreads();
  }
  if (tid == 0) { scalars[0] = loss; scalars[1] = perp; }
}

extern "C" void kernel_launch(void* const* d_in, const int* in_sizes, int n_in,
                              void* d_out, int out_size, void* d_ws, size_t ws_size,
                              hipStream_t stream) {
  (void)in_sizes; (void)n_in; (void)out_size; (void)ws_size;
  const float* z   = (const float*)d_in[0];                 // [N_TOK, DIM]
  const float* cbs = (const float*)d_in[1];                 // [NCB, KCODES, DIM]
  float* out = (float*)d_out;

  char* ws = (char*)d_ws;
  float* res       = (float*)ws;  ws += (size_t)N_TOK * DIM * sizeof(float);        // 32 MB
  bf16*  cbbf      = (bf16*) ws;  ws += (size_t)NCB * KCODES * DIM * sizeof(bf16);  // 8 MB
  float* cbn       = (float*)ws;  ws += (size_t)NCB * KCODES * sizeof(float);       // 64 KB
  int*   idx       = (int*)  ws;  ws += (size_t)N_TOK * sizeof(int);                // 128 KB
  int*   hist      = (int*)  ws;  ws += (size_t)NCB * KCODES * sizeof(int);         // 64 KB
  float* tok_sumsq = (float*)ws;  ws += (size_t)NCB * N_TOK * sizeof(float);        // 512 KB

  float* qsum    = out;
  float* idxf    = out + (size_t)N_TOK * DIM;
  float* scalars = idxf + (size_t)NCB * N_TOK;

  hipMemcpyAsync(res, z, (size_t)N_TOK * DIM * sizeof(float),
                 hipMemcpyDeviceToDevice, stream);
  rqvae_zero_hist<<<(NCB * KCODES + 255) / 256, 256, 0, stream>>>(hist);
  rqvae_prep_cb<<<NCB * KCODES, 256, 0, stream>>>(cbs, cbbf, cbn);

  for (int l = 0; l < NCB; ++l) {
    rqvae_argmin_wmma<<<N_TOK / 128, 128, 0, stream>>>(
        res, cbbf + (size_t)l * KCODES * DIM, cbn + (size_t)l * KCODES, idx);
    rqvae_update<<<N_TOK, 256, 0, stream>>>(
        cbs + (size_t)l * KCODES * DIM, idx, res, qsum,
        idxf + (size_t)l * N_TOK, hist + (size_t)l * KCODES,
        tok_sumsq + (size_t)l * N_TOK, l == 0 ? 1 : 0);
  }

  rqvae_finalize<<<1, 256, 0, stream>>>(hist, tok_sumsq, scalars);
}
